// CrossAttention_71863392796763
// MI455X (gfx1250) — matrix operand order repaired and checked by reading, not compiled, not run on hardware
//
#include <hip/hip_runtime.h>
#include <hip/hip_bf16.h>
#include <stdint.h>

#define B_   2
#define SQ_  2048
#define SK_  2048
#define H_   32
#define HK_  8
#define D_   128
#define GQA_ (H_ / HK_)   // 4 query heads per kv head

typedef __attribute__((ext_vector_type(16))) __bf16 v16bf;
typedef __attribute__((ext_vector_type(8)))  __bf16 v8bf;
typedef __attribute__((ext_vector_type(4)))  __bf16 v4bf;
typedef __attribute__((ext_vector_type(8)))  float  v8f;
typedef __attribute__((ext_vector_type(4)))  uint32_t u32x4;
typedef __attribute__((ext_vector_type(4)))  int      i32x4;
typedef __attribute__((ext_vector_type(8)))  int      i32x8;

// Toolchain discrimination for the TDM builtin (5-arg ROCm vs 6-arg therock)
#if defined(__has_include)
#if __has_include(<hip/amd_detail/amd_gfx1250_TDM.h>)
#define TDM_6ARG 1
#else
#define TDM_6ARG 0
#endif
#else
#define TDM_6ARG 0
#endif

#if defined(__has_builtin)
#if __has_builtin(__builtin_amdgcn_tensor_load_to_lds)
#define HAVE_TDM 1
#else
#define HAVE_TDM 0
#endif
#else
#define HAVE_TDM 0
#endif

// bf16 KV scratch layout in d_ws:
//   Kbf[b][hk][s][d]  : B*HK*SK*D bf16
//   Vt [b][hk][d][s]  : B*HK*D*SK bf16   (pre-transposed)
//   lengths[B]        : int
#define KBF_BYTES ((size_t)B_ * HK_ * SK_ * D_ * 2)

// ---------------------------------------------------------------------------
// Pre-kernel 1: lengths[b] = sum(key_padding_mask[b, :])
// ---------------------------------------------------------------------------
__global__ void ca_len_kernel(const unsigned char* __restrict__ mask,
                              int* __restrict__ lengths) {
  __shared__ int red[256];
  const int b = blockIdx.x, t = threadIdx.x;
  int s = 0;
  for (int i = t; i < SK_; i += 256)
    s += (mask[(size_t)b * SK_ + i] != 0) ? 1 : 0;
  red[t] = s;
  __syncthreads();
  for (int o = 128; o > 0; o >>= 1) {
    if (t < o) red[t] += red[t + o];
    __syncthreads();
  }
  if (t == 0) lengths[b] = red[0];
}

// ---------------------------------------------------------------------------
// Pre-kernel 2: one-time fp32 -> bf16 conversion of KV; V transposed so both
// LDS tiles in the main loop are plain 2-D row-major copies (TDM-friendly).
// grid = (SK/64, HK, B), block = 256.
// ---------------------------------------------------------------------------
__global__ __launch_bounds__(256) void ca_cvt_kernel(
    const float* __restrict__ kv,
    __bf16* __restrict__ kbf,
    __bf16* __restrict__ vt) {
  __shared__ __align__(16) __bf16 Vtl[128 * 64];
  const int t  = threadIdx.x;
  const int kb = blockIdx.x * 64;
  const int hk = blockIdx.y;
  const int b  = blockIdx.z;
  const size_t bh = (size_t)b * HK_ + hk;
#pragma unroll
  for (int i = 0; i < 8; ++i) {
    const int lin = t + 256 * i;        // 0..2047
    const int key = lin >> 5;           // 0..63
    const int d4  = (lin & 31) << 2;    // 0..124
    const int s   = kb + key;
    const size_t kidx = ((((size_t)b * SK_ + s) * 2 + 0) * HK_ + hk) * (size_t)D_ + d4;
    const size_t vidx = ((((size_t)b * SK_ + s) * 2 + 1) * HK_ + hk) * (size_t)D_ + d4;
    const float4 fk = *(const float4*)(kv + kidx);
    const float4 fv = *(const float4*)(kv + vidx);
    v4bf pk = {(__bf16)fk.x, (__bf16)fk.y, (__bf16)fk.z, (__bf16)fk.w};
    *(v4bf*)&kbf[(bh * SK_ + s) * D_ + d4] = pk;
    Vtl[(d4 + 0) * 64 + key] = (__bf16)fv.x;
    Vtl[(d4 + 1) * 64 + key] = (__bf16)fv.y;
    Vtl[(d4 + 2) * 64 + key] = (__bf16)fv.z;
    Vtl[(d4 + 3) * 64 + key] = (__bf16)fv.w;
  }
  __syncthreads();
#pragma unroll
  for (int i = 0; i < 8; ++i) {
    const int lin = t + 256 * i;        // 0..2047
    const int row = lin >> 4;           // d, 0..127
    const int c4  = (lin & 15) << 2;    // key chunk 0..60
    *(v4bf*)&vt[(bh * D_ + row) * SK_ + kb + c4] = *(const v4bf*)&Vtl[row * 64 + c4];
  }
}

#if HAVE_TDM
// Issue one 2-D TENSOR_LOAD_TO_LDS: tile (tile1 rows x tile0 elems), bf16.
// D# per CDNA5 ISA.
__device__ __forceinline__ void tdm_load_2d(uint32_t lds_off, uint64_t gaddr,
                                            uint32_t td0, uint32_t td1,
                                            uint32_t tile0, uint32_t tile1,
                                            uint32_t stride0) {
  u32x4 g0 = {0x1u,                     // count=1, is_restore=0, gather=0
              lds_off,                  // lds_addr (bytes)
              (uint32_t)gaddr,          // global_addr[31:0]
              (uint32_t)((gaddr >> 32) & 0x01FFFFFFu) | 0x80000000u}; // addr[56:32] | type=2
  i32x8 g1 = {(int)0x00010000u,                         // wg_mask=0, data_size=1 (2B)
              (int)((td0 & 0xFFFFu) << 16),             // abar=0 | tensor_dim0 lo
              (int)((td0 >> 16) | ((td1 & 0xFFFFu) << 16)),
              (int)((td1 >> 16) | (tile0 << 16)),
              (int)tile1,                               // tile_dim1 | tile_dim2=0
              (int)stride0,                             // dim0_stride lo
              0, 0};
  i32x4 z4 = {0, 0, 0, 0};
#if TDM_6ARG
  i32x8 z8 = {0, 0, 0, 0, 0, 0, 0, 0};
  __builtin_amdgcn_tensor_load_to_lds(g0, g1, z4, z4, z8, 0);
#else
  __builtin_amdgcn_tensor_load_to_lds(g0, g1, z4, z4, 0);
#endif
}
#endif

// ---------------------------------------------------------------------------
// Flash-attention main kernel. grid = (SQ/32, HK, B), block = 256 (8 waves).
// Double-buffered K/V LDS tiles; TDM (or fallback vector copies) prefetches
// tile n+1 while waves compute on tile n.
// ---------------------------------------------------------------------------
__global__ __launch_bounds__(256) void ca_attn_kernel(
    const float* __restrict__ q,
    const __bf16* __restrict__ kbf,
    const __bf16* __restrict__ vt,
    const unsigned char* __restrict__ mask,
    const int* __restrict__ lengths,
    float* __restrict__ out) {
  __shared__ __align__(16) __bf16 Klds[2][64 * 128];   // [key][d]
  __shared__ __align__(16) __bf16 Vtlds[2][128 * 64];  // [d][key]
  __shared__ __align__(16) __bf16 Plds[8][16 * 64];    // per-wave P scratch
  __shared__ float padb[2][64];

  const int t    = threadIdx.x;
  const int lane = t & 31;
  const int w    = t >> 5;
  const int half = lane >> 4;
  const int ncol = lane & 15;
  const int b    = blockIdx.z;
  const int hk   = blockIdx.y;
  const int h    = hk * GQA_ + (w & 3);
  const int qbase = blockIdx.x * 32 + (w >> 2) * 16;
  const int skv  = lengths[b];
  const size_t bh = (size_t)b * HK_ + hk;
  const float scale = 0.08838834764831845f; // 1/sqrt(128)

  // ---- Preload Q as 4 bf16 A-fragments ----
  v16bf qf[4];
  {
    const int qrow = qbase + ncol;
    const float* qp = q + (((size_t)b * SQ_ + qrow) * H_ + h) * D_;
#pragma unroll
    for (int c = 0; c < 4; ++c) {
      const int d0 = c * 32 + half * 8;
      const int d1 = c * 32 + 16 + half * 8;
      float4 a0 = *(const float4*)(qp + d0);
      float4 a1 = *(const float4*)(qp + d0 + 4);
      float4 b0 = *(const float4*)(qp + d1);
      float4 b1 = *(const float4*)(qp + d1 + 4);
      v16bf f;
      f[0] = (__bf16)a0.x; f[1] = (__bf16)a0.y; f[2]  = (__bf16)a0.z; f[3]  = (__bf16)a0.w;
      f[4] = (__bf16)a1.x; f[5] = (__bf16)a1.y; f[6]  = (__bf16)a1.z; f[7]  = (__bf16)a1.w;
      f[8] = (__bf16)b0.x; f[9] = (__bf16)b0.y; f[10] = (__bf16)b0.z; f[11] = (__bf16)b0.w;
      f[12] = (__bf16)b1.x; f[13] = (__bf16)b1.y; f[14] = (__bf16)b1.z; f[15] = (__bf16)b1.w;
      qf[c] = f;
    }
  }

  const v8f vzero = {0, 0, 0, 0, 0, 0, 0, 0};
  v8f O[8];
#pragma unroll
  for (int i = 0; i < 8; ++i) O[i] = vzero;
  float mrow[8], lsum[8];
#pragma unroll
  for (int r = 0; r < 8; ++r) { mrow[r] = -3.0e38f; lsum[r] = 0.0f; }

  // ---- Prologue: fetch tile 0 into buffer 0 ----
#if HAVE_TDM
  if (w == 0) {
    tdm_load_2d((uint32_t)(uintptr_t)&Klds[0][0],
                (uint64_t)(uintptr_t)&kbf[(bh * SK_ + 0) * D_],
                D_, 64, D_, 64, D_);
    tdm_load_2d((uint32_t)(uintptr_t)&Vtlds[0][0],
                (uint64_t)(uintptr_t)&vt[(bh * D_ + 0) * SK_ + 0],
                64, D_, 64, D_, SK_);
  }
#else
#pragma unroll
  for (int i = 0; i < 8; ++i) {
    const int lin = t + 256 * i;
    const int key = lin >> 5, d4 = (lin & 31) << 2;
    *(v4bf*)&Klds[0][key * 128 + d4] = *(const v4bf*)&kbf[(bh * SK_ + key) * D_ + d4];
    const int row = lin >> 4, c4 = (lin & 15) << 2;
    *(v4bf*)&Vtlds[0][row * 64 + c4] = *(const v4bf*)&vt[(bh * D_ + row) * SK_ + c4];
  }
#endif
  if (t < 64) padb[0][t] = mask[(size_t)b * SK_ + t] ? 0.0f : -10000.0f;

  for (int kb = 0; kb < SK_; kb += 64) {
    const int buf = (kb >> 6) & 1;
#if HAVE_TDM
    if (w == 0) __builtin_amdgcn_s_wait_tensorcnt((short)0);
#endif
    __syncthreads();  // tile[buf] ready for all; tile[buf^1] fully consumed

    // ---- Prefetch next tile into the other buffer (overlaps with compute) ----
    if (kb + 64 < SK_) {
      const int nb = kb + 64;
#if HAVE_TDM
      if (w == 0) {
        tdm_load_2d((uint32_t)(uintptr_t)&Klds[buf ^ 1][0],
                    (uint64_t)(uintptr_t)&kbf[(bh * SK_ + nb) * D_],
                    D_, 64, D_, 64, D_);
        tdm_load_2d((uint32_t)(uintptr_t)&Vtlds[buf ^ 1][0],
                    (uint64_t)(uintptr_t)&vt[(bh * D_ + 0) * SK_ + nb],
                    64, D_, 64, D_, SK_);
      }
#else
#pragma unroll
      for (int i = 0; i < 8; ++i) {
        const int lin = t + 256 * i;
        const int key = lin >> 5, d4 = (lin & 31) << 2;
        *(v4bf*)&Klds[buf ^ 1][key * 128 + d4] =
            *(const v4bf*)&kbf[(bh * SK_ + nb + key) * D_ + d4];
        const int row = lin >> 4, c4 = (lin & 15) << 2;
        *(v4bf*)&Vtlds[buf ^ 1][row * 64 + c4] =
            *(const v4bf*)&vt[(bh * D_ + row) * SK_ + nb + c4];
      }
#endif
      if (t < 64)
        padb[buf ^ 1][t] = mask[(size_t)b * SK_ + nb + t] ? 0.0f : -10000.0f;
    }

    const __bf16* Kb = &Klds[buf][0];
    const __bf16* Vb = &Vtlds[buf][0];

    // ---- S = Q K^T ----
    v8f S[4];
#pragma unroll
    for (int nt = 0; nt < 4; ++nt) {
      v8f acc = vzero;
#pragma unroll
      for (int c = 0; c < 4; ++c) {
        const v16bf bk = __builtin_shufflevector(*(const v8bf*)&Kb[(nt * 16 + ncol) * 128 + c * 32 + half * 8], *(const v8bf*)&Kb[(nt * 16 + ncol) * 128 + c * 32 + 16 + half * 8], 0, 1, 2, 3, 4, 5, 6, 7, 8, 9, 10, 11, 12, 13, 14, 15);
        acc = __builtin_amdgcn_wmma_f32_16x16x32_bf16(
            false, qf[c], false, bk, (short)0, acc, false, false);
      }
      S[nt] = acc;
    }

    // ---- scale + pad bias + causal override (reference order) ----
#pragma unroll
    for (int nt = 0; nt < 4; ++nt) {
      const float pb  = padb[buf][nt * 16 + ncol];
      const int  kcol = kb + nt * 16 + ncol;
#pragma unroll
      for (int r = 0; r < 8; ++r) {
        const int mr = qbase + r + 8 * half;
        float s = S[nt][r] * scale + pb;
        if (kcol > mr + skv - SQ_) s = -10000.0f;
        S[nt][r] = s;
      }
    }

    // ---- online softmax ----
    float alpha[8];
#pragma unroll
    for (int r = 0; r < 8; ++r) {
      float m0 = fmaxf(fmaxf(S[0][r], S[1][r]), fmaxf(S[2][r], S[3][r]));
#pragma unroll
      for (int off = 8; off >= 1; off >>= 1)
        m0 = fmaxf(m0, __shfl_xor(m0, off, 16));
      const float mn = fmaxf(mrow[r], m0);
      alpha[r] = __expf(mrow[r] - mn);
      mrow[r] = mn;
    }
#pragma unroll
    for (int r = 0; r < 8; ++r) {
      float s0 = 0.0f;
#pragma unroll
      for (int nt = 0; nt < 4; ++nt) {
        const float p = __expf(S[nt][r] - mrow[r]);
        S[nt][r] = p;
        s0 += p;
      }
#pragma unroll
      for (int off = 8; off >= 1; off >>= 1)
        s0 += __shfl_xor(s0, off, 16);
      lsum[r] = lsum[r] * alpha[r] + s0;
    }
#pragma unroll
    for (int dt = 0; dt < 8; ++dt)
#pragma unroll
      for (int r = 0; r < 8; ++r) O[dt][r] *= alpha[r];

    // ---- P (C layout) -> per-wave LDS -> A-fragments ----
    __bf16* pl = &Plds[w][0];
#pragma unroll
    for (int nt = 0; nt < 4; ++nt)
#pragma unroll
      for (int r = 0; r < 8; ++r)
        pl[(r + 8 * half) * 64 + nt * 16 + ncol] = (__bf16)S[nt][r];
    asm volatile("s_wait_dscnt 0" ::: "memory");  // wave-internal RAW via LDS

    // ---- O += P V ----
#pragma unroll
    for (int a = 0; a < 2; ++a) {
      const v8bf lo = *(const v8bf*)&pl[ncol * 64 + a * 32 + half * 8];
      const v8bf hi = *(const v8bf*)&pl[ncol * 64 + a * 32 + 16 + half * 8];
      v16bf ap;
#pragma unroll
      for (int i = 0; i < 8; ++i) { ap[i] = lo[i]; ap[8 + i] = hi[i]; }
#pragma unroll
      for (int dt = 0; dt < 8; ++dt) {
        const v16bf bv = __builtin_shufflevector(*(const v8bf*)&Vb[(dt * 16 + ncol) * 64 + a * 32 + half * 8], *(const v8bf*)&Vb[(dt * 16 + ncol) * 64 + a * 32 + 16 + half * 8], 0, 1, 2, 3, 4, 5, 6, 7, 8, 9, 10, 11, 12, 13, 14, 15);
        O[dt] = __builtin_amdgcn_wmma_f32_16x16x32_bf16(
            false, ap, false, bv, (short)0, O[dt], false, false);
      }
    }
  }

  // ---- normalize and store ----
#pragma unroll
  for (int r = 0; r < 8; ++r) {
    const float inv = 1.0f / lsum[r];
    const int row = qbase + r + 8 * half;
    float* op = out + (((size_t)b * SQ_ + row) * H_ + h) * D_;
#pragma unroll
    for (int dt = 0; dt < 8; ++dt)
      op[dt * 16 + ncol] = O[dt][r] * inv;
  }
}

// ---------------------------------------------------------------------------
extern "C" void kernel_launch(void* const* d_in, const int* in_sizes, int n_in,
                              void* d_out, int out_size, void* d_ws, size_t ws_size,
                              hipStream_t stream) {
  const float* q  = (const float*)d_in[0];
  const float* kv = (const float*)d_in[1];
  const unsigned char* mask = (const unsigned char*)d_in[2];
  float* out = (float*)d_out;

  __bf16* kbf   = (__bf16*)d_ws;
  __bf16* vt    = (__bf16*)((char*)d_ws + KBF_BYTES);
  int* lengths  = (int*)((char*)d_ws + 2 * KBF_BYTES);

  ca_len_kernel<<<B_, 256, 0, stream>>>(mask, lengths);

  dim3 cgrid(SK_ / 64, HK_, B_);
  ca_cvt_kernel<<<cgrid, 256, 0, stream>>>(kv, kbf, vt);

  dim3 grid(SQ_ / 32, HK_, B_);
  ca_attn_kernel<<<grid, 256, 0, stream>>>(q, kbf, vt, mask, lengths, out);
}
